// RelativePositionalEncoding3D_16492674417481
// MI455X (gfx1250) — matrix operand order repaired and checked by reading, not compiled, not run on hardware
//
#include <hip/hip_runtime.h>
#include <hip/hip_bf16.h>

typedef __attribute__((ext_vector_type(16))) _Float16 v16h;
typedef __attribute__((ext_vector_type(8)))  float    v8f;
typedef __attribute__((ext_vector_type(4)))  float    v4f;
typedef __attribute__((ext_vector_type(4)))  int      v4i;

#define T_     50
#define HW_    64
#define NH_    16
#define HID_   512
#define ROWS_  22275   // (2T-1)*(2H-1)*(2W-1) = 99*15*15
#define NTILES_ 1393   // ceil(ROWS_/16)

__device__ __forceinline__ float fast_scaled_sigmoid(float x) {
  // 16 / (1 + exp(-x)) with hardware rcp (no IEEE div sequence needed)
  return 16.0f * __builtin_amdgcn_rcpf(1.0f + __expf(-x));
}

// ---------------------------------------------------------------------------
// Kernel A: srpt[row][nh] = 16*sigmoid( (relu(rct_row @ W1 + b1) @ W2)[nh] )
// One wave per 16-row tile; K=512 contraction via 16x v_wmma_f32_16x16x32_f16.
// ---------------------------------------------------------------------------
__global__ __launch_bounds__(128) void cpb_mlp_wmma_kernel(
    const float* __restrict__ W1, const float* __restrict__ b1,
    const float* __restrict__ W2, const float* __restrict__ rct,
    float* __restrict__ srpt)
{
  __shared__ float4   sW1[HID_];        // {W1[0][j], W1[1][j], W1[2][j], b1[j]}
  __shared__ _Float16 sW2t[NH_ * 514];  // W2 transposed [n][k], padded stride

  for (int j = threadIdx.x; j < HID_; j += 128)
    sW1[j] = make_float4(W1[j], W1[HID_ + j], W1[2 * HID_ + j], b1[j]);
  for (int t = threadIdx.x; t < HID_ * NH_; t += 128) {
    int k = t >> 4, n = t & 15;                 // coalesced global read of W2
    sW2t[n * 514 + k] = (_Float16)W2[k * NH_ + n];
  }
  __syncthreads();

  const int wave = threadIdx.x >> 5;
  const int lane = threadIdx.x & 31;
  const int tile = blockIdx.x * 4 + wave;       // wave-uniform guard
  if (tile >= NTILES_) return;

  const int rowBase = tile * 16;
  const int n = lane & 15;      // matrix column / row-in-tile selector
  const int g = lane >> 4;      // half-wave group
  int m = rowBase + n;          // this lane's A-matrix row (clamped, no branch)
  if (m > ROWS_ - 1) m = ROWS_ - 1;
  const float r0 = rct[m * 3 + 0];
  const float r1 = rct[m * 3 + 1];
  const float r2 = rct[m * 3 + 2];

  const unsigned* __restrict__ w2u = (const unsigned*)(&sW2t[n * 514]);

  v8f acc = {};
  for (int kb = 0; kb < 16; ++kb) {             // K blocks of 32
    union { v16h v; _Float16 e[16]; } af;
    union { v16h v; unsigned  u[8];  } bf;
    #pragma unroll
    for (int vv = 0; vv < 8; ++vv) {
      #pragma unroll
      for (int h = 0; h < 2; ++h) {
        // ISA 16-bit A layout: lane L (g=L/16), VGPR v holds
        // K = 16*(v>=4) + 8*g + 2*(v&3) + h
        int j = kb * 32 + ((vv >= 4) ? 16 : 0) + 8 * g + 2 * (vv & 3) + h;
        float4 w = sW1[j];
        float hv = fmaf(r0, w.x, fmaf(r1, w.y, fmaf(r2, w.z, w.w)));
        af.e[2 * vv + h] = (_Float16)fmaxf(hv, 0.0f);   // relu + cvt
      }
      // ISA 16-bit B layout: lane column n, VGPR v holds K = 16*(v>=4) + 8*g + 2*(v&3) {+1}
      int k = kb * 32 + ((vv >= 4) ? 16 : 0) + 8 * g + 2 * (vv & 3);        // even -> aligned b32
      bf.u[vv] = w2u[k >> 1];
    }
    acc = __builtin_amdgcn_wmma_f32_16x16x32_f16(
        false, af.v, false, bf.v, (short)0, acc, false, false);
  }

  // C/D layout: VGPR i -> M = i + 8*g, N = n. Fold RPE_COEF * sigmoid here.
  float* __restrict__ dst = srpt + (rowBase + 8 * g) * NH_ + n;
  if (rowBase + 16 <= ROWS_) {
    // full tile (all but the last): branch-free epilogue
    #pragma unroll
    for (int i = 0; i < 8; ++i)
      dst[i * NH_] = fast_scaled_sigmoid(acc[i]);
  } else {
    #pragma unroll
    for (int i = 0; i < 8; ++i)
      if (rowBase + i + 8 * g < ROWS_)
        dst[i * NH_] = fast_scaled_sigmoid(acc[i]);
  }
}

// ---------------------------------------------------------------------------
// Kernel B: out[b][nh][F][P] = srpt[ rpi[f[b][F/64]][F%64][-p[b][P/64]][P%64] ][nh]
// Each thread: 4 consecutive P -> 1x b128 index load, 16x b128 srpt gathers,
// 16x nontemporal b128 coalesced stores. Pure HBM-roofline streaming.
// ---------------------------------------------------------------------------
__global__ __launch_bounds__(256) void rpe_gather_kernel(
    const int* __restrict__ ptc, const int* __restrict__ ftc,
    const int* __restrict__ rpi, const float* __restrict__ srpt,
    float* __restrict__ out)
{
  int tid = blockIdx.x * 256 + threadIdx.x;     // total = 8*512*128
  int p4  = tid & 127;
  int F   = (tid >> 7) & 511;
  int b   = tid >> 16;
  int P0  = p4 << 2;

  int tf = F >> 6,  hwf  = F  & 63;
  int tp = P0 >> 6, hwp0 = P0 & 63;             // same tp for all 4 P's
  int t_f =  ftc[b * 8 + tf];
  int t_p = -ptc[b * 8 + tp];

  long ibase = ((long)(t_f * HW_ + hwf) * T_ + t_p) * HW_ + hwp0;
  v4i idx = *(const v4i*)(rpi + ibase);         // 16B aligned (hwp0 % 4 == 0)

  v4f q[4][4];
  #pragma unroll
  for (int j = 0; j < 4; ++j) {
    const v4f* __restrict__ src = (const v4f*)(srpt + (long)idx[j] * NH_);
    q[j][0] = src[0]; q[j][1] = src[1]; q[j][2] = src[2]; q[j][3] = src[3];
  }

  long obase = ((long)(b * NH_) * 512 + F) * 512 + P0;
  #pragma unroll
  for (int nh = 0; nh < NH_; ++nh) {
    v4f o;
    o[0] = q[0][nh >> 2][nh & 3];
    o[1] = q[1][nh >> 2][nh & 3];
    o[2] = q[2][nh >> 2][nh & 3];
    o[3] = q[3][nh >> 2][nh & 3];
    __builtin_nontemporal_store(o, (v4f*)(out + obase + (long)nh * 512 * 512));
  }
}

extern "C" void kernel_launch(void* const* d_in, const int* in_sizes, int n_in,
                              void* d_out, int out_size, void* d_ws, size_t ws_size,
                              hipStream_t stream) {
  const int*   ptc = (const int*)  d_in[0];
  const int*   ftc = (const int*)  d_in[1];
  const float* W1  = (const float*)d_in[2];
  const float* b1  = (const float*)d_in[3];
  const float* W2  = (const float*)d_in[4];
  const float* rct = (const float*)d_in[5];
  const int*   rpi = (const int*)  d_in[6];
  float* out  = (float*)d_out;
  float* srpt = (float*)d_ws;                   // 22275*16*4 = 1.43 MB scratch

  cpb_mlp_wmma_kernel<<<(NTILES_ + 3) / 4, 128, 0, stream>>>(W1, b1, W2, rct, srpt);
  rpe_gather_kernel<<<(8 * 512 * 128) / 256, 256, 0, stream>>>(ptc, ftc, rpi, srpt, out);
}
